// Encoder_10599979287062
// MI455X (gfx1250) — matrix operand order repaired and checked by reading, not compiled, not run on hardware
//
#include <hip/hip_runtime.h>

typedef __attribute__((ext_vector_type(16))) _Float16 v16h;
typedef __attribute__((ext_vector_type(8)))  float    v8f;

#define B_   2
#define L_   2048
#define H_   12
#define D_   64
#define MQ   32                 // queries per workgroup (2 x 16-row subtiles)
#define NTHREADS 512
#define NWAVES 16
#define KPW  (L_ / NWAVES)      // 128 keys per wave
#define SSTRIDE (L_ + 8)        // padded score row stride (floats); 16B-aligned
#define ROWS_ (H_ * D_)         // 768 floats between consecutive seq rows
#define VT_BYTES ((size_t)B_ * H_ * D_ * L_ * 2)   // f16 V-transpose workspace

// ---- fragment builders -----------------------------------------------------
// A fragment 16x32 as built here: lane l -> m = l&15, koffA = (l>>4)*16;
// element e = K[koffA+e] (contiguous k order, same as B).
__device__ inline v16h frag_a_f32(const float* __restrict__ p, float scale) {
  v16h r;
#pragma unroll
  for (int blk = 0; blk < 2; ++blk) {
    const float4* p4 = (const float4*)(p + blk * 8);
    float4 f0 = p4[0], f1 = p4[1];
    r[8*blk+0] = (_Float16)(f0.x*scale); r[8*blk+1] = (_Float16)(f0.y*scale);
    r[8*blk+2] = (_Float16)(f0.z*scale); r[8*blk+3] = (_Float16)(f0.w*scale);
    r[8*blk+4] = (_Float16)(f1.x*scale); r[8*blk+5] = (_Float16)(f1.y*scale);
    r[8*blk+6] = (_Float16)(f1.z*scale); r[8*blk+7] = (_Float16)(f1.w*scale);
  }
  return r;
}

// Same A layout; values are poly weights computed from LDS scores (float4 reads).
__device__ inline v16h frag_a_w(const float* __restrict__ s, float cm) {
  const float4* s4 = (const float4*)s;
  float4 f0 = s4[0], f1 = s4[1];   // s[0..7]
  float4 f2 = s4[2], f3 = s4[3];   // s[8..15]
  float x[16] = { f0.x,f0.y,f0.z,f0.w, f1.x,f1.y,f1.z,f1.w,
                  f2.x,f2.y,f2.z,f2.w, f3.x,f3.y,f3.z,f3.w };
  v16h r;
#pragma unroll
  for (int i = 0; i < 16; ++i) {
    float y = 1.0f / (-x[i] - cm);
    r[i] = (_Float16)(y * y);
  }
  return r;
}

// B fragment 32x16 as built here: lane l -> n = l&15, koffB = (l>>4)*16;
// element e = K[koffB+e].  p pre-offset so p[e] = B[koffB+e][n].
__device__ inline v16h frag_b_f32(const float* __restrict__ p) {
  v16h r;
#pragma unroll
  for (int i = 0; i < 4; ++i) {
    float4 f = ((const float4*)p)[i];
    r[4*i+0] = (_Float16)f.x; r[4*i+1] = (_Float16)f.y;
    r[4*i+2] = (_Float16)f.z; r[4*i+3] = (_Float16)f.w;
  }
  return r;
}

__device__ inline v8f wmma16(v16h a, v16h b, v8f c) {
  return __builtin_amdgcn_wmma_f32_16x16x32_f16(false, a, false, b,
                                                (short)0, c, false, false);
}

// ---- pre-pass: Vt[b][h][d][l] (f16) = V[b][l][h][d] -------------------------
__global__ __launch_bounds__(256)
void transpose_v_kernel(const float* __restrict__ Vg, unsigned* __restrict__ Vt) {
  const int idx = blockIdx.x * 256 + threadIdx.x;    // one output dword (2 l's)
  const int lp = idx & 1023;                         // l-pair
  const int d  = (idx >> 10) & 63;
  const int bh = idx >> 16;                          // b*H + h
  const int h  = bh % H_;
  const int b  = bh / H_;
  const size_t base = (((size_t)b * L_ + 2 * lp) * H_ + h) * D_ + d;
  union { _Float16 h2[2]; unsigned u; } pk;
  pk.h2[0] = (_Float16)Vg[base];
  pk.h2[1] = (_Float16)Vg[base + ROWS_];
  Vt[idx] = pk.u;
}

// ---------------------------------------------------------------------------
__global__ __launch_bounds__(NTHREADS)
void poly_attn_kernel(const float* __restrict__ Qg,
                      const float* __restrict__ Kg,
                      const float* __restrict__ Vg,
                      const _Float16* __restrict__ Vt,   // may be nullptr
                      float* __restrict__ Og) {
  __shared__ float S[MQ][SSTRIDE];     // score rows (f32), ~263 KB
  __shared__ float cval[MQ];
  __shared__ float Ored[MQ][D_];       // 8 KB output reduction buffer

  const int tid  = threadIdx.x;
  const int wave = tid >> 5;
  const int lane = tid & 31;
  const int lo   = lane & 15;
  const int hi   = lane >> 4;

  const int wg    = blockIdx.x;
  const int qtile = wg & 63;               // L_/MQ = 64
  const int h     = (wg >> 6) % H_;
  const int b     = (wg >> 6) / H_;
  const int q0    = qtile * MQ;

  const float* Qb = Qg + (size_t)b * L_ * ROWS_ + h * D_;
  const float* Kb = Kg + (size_t)b * L_ * ROWS_ + h * D_;
  const float* Vb = Vg + (size_t)b * L_ * ROWS_ + h * D_;
  float*       Ob = Og + (size_t)b * L_ * ROWS_ + h * D_;

  // ---- phase 1: S = (Q/8) * K^T over this wave's 128-key slab -------------
  {
    v16h aQ0[2], aQ1[2];
#pragma unroll
    for (int sub = 0; sub < 2; ++sub) {
      const float* qrow = Qb + (size_t)(q0 + sub * 16 + lo) * ROWS_ + hi * 16;
      aQ0[sub] = frag_a_f32(qrow,      0.125f);   // d = 0..31
      aQ1[sub] = frag_a_f32(qrow + 32, 0.125f);   // d = 32..63
    }
#pragma unroll 2
    for (int t = 0; t < KPW / 16; ++t) {          // 8 key tiles of 16
      const int n0 = wave * KPW + t * 16;
      const float* krow = Kb + (size_t)(n0 + lo) * ROWS_ + hi * 16;
      v16h bK0 = frag_b_f32(krow);        // d 0..31 slice
      v16h bK1 = frag_b_f32(krow + 32);   // d 32..63 slice
#pragma unroll
      for (int sub = 0; sub < 2; ++sub) { // subtiles share K frags
        v8f acc = {0.f,0.f,0.f,0.f,0.f,0.f,0.f,0.f};
        acc = wmma16(aQ0[sub], bK0, acc);
        acc = wmma16(aQ1[sub], bK1, acc);
#pragma unroll
        for (int j = 0; j < 8; ++j)
          S[sub * 16 + j + 8 * hi][n0 + lo] = acc[j];
      }
    }
  }
  __syncthreads();

  // ---- phase 2: Newton solve (one 16-lane group per score row) ------------
  {
    const int r = tid >> 4;
    const int i = tid & 15;                  // float4 columns i*4 + 64*jj
    const float4* Srow4 = (const float4*)&S[r][0];
    float pm = -3.4e38f;
#pragma unroll 4
    for (int jj = 0; jj < L_ / 64; ++jj) {
      float4 f = Srow4[i + 16 * jj];
      pm = fmaxf(pm, fmaxf(fmaxf(f.x, f.y), fmaxf(f.z, f.w)));
    }
#pragma unroll
    for (int m = 8; m >= 1; m >>= 1)
      pm = fmaxf(pm, __shfl_xor(pm, m, 16));
    float c = -pm - 1.0f;
    for (int it = 0; it < 6; ++it) {
      float s2 = 0.f, s3 = 0.f;
#pragma unroll 4
      for (int jj = 0; jj < L_ / 64; ++jj) {
        float4 f = Srow4[i + 16 * jj];
        float y0 = 1.0f / (-f.x - c), y1 = 1.0f / (-f.y - c);
        float y2 = 1.0f / (-f.z - c), y3 = 1.0f / (-f.w - c);
        float a0 = y0*y0, a1 = y1*y1, a2 = y2*y2, a3 = y3*y3;
        s2 += (a0 + a1) + (a2 + a3);
        s3 += (a0*y0 + a1*y1) + (a2*y2 + a3*y3);
      }
#pragma unroll
      for (int m = 8; m >= 1; m >>= 1) {
        s2 += __shfl_xor(s2, m, 16);
        s3 += __shfl_xor(s3, m, 16);
      }
      c = c - (s2 - 1.0f) / (2.0f * s3 + 1e-8f);
    }
    if (i == 0) cval[r] = c;
  }

  // zero output reduction buffer
#pragma unroll
  for (int idx = tid; idx < MQ * D_; idx += NTHREADS)
    ((float*)Ored)[idx] = 0.0f;
  __syncthreads();   // cval + Ored ready; S stable

  // ---- phase 3: O += W * V over this wave's 128-key slab ------------------
  v8f accO[2][4];
#pragma unroll
  for (int s = 0; s < 2; ++s)
#pragma unroll
    for (int j = 0; j < 4; ++j)
      accO[s][j] = (v8f){0.f,0.f,0.f,0.f,0.f,0.f,0.f,0.f};

  {
    const float cm0 = cval[lo];
    const float cm1 = cval[16 + lo];
    if (Vt != nullptr) {
      // fast path: contiguous 32B loads from transposed f16 V
      const _Float16* Vtb = Vt + (size_t)(b * H_ + h) * D_ * L_;
#pragma unroll 2
      for (int kc = 0; kc < KPW / 32; ++kc) {
        const int kbase = wave * KPW + kc * 32;
        v16h aW0 = frag_a_w(&S[lo][kbase + hi * 16],      cm0);
        v16h aW1 = frag_a_w(&S[16 + lo][kbase + hi * 16], cm1);
#pragma unroll
        for (int j = 0; j < 4; ++j) {
          const _Float16* vp =
              Vtb + (size_t)(j * 16 + lo) * L_ + kbase + hi * 16;
          v16h bV = *(const v16h*)vp;
          accO[0][j] = wmma16(aW0, bV, accO[0][j]);
          accO[1][j] = wmma16(aW1, bV, accO[1][j]);
        }
      }
    } else {
      // fallback: strided f32 gathers from V
#pragma unroll 2
      for (int kc = 0; kc < KPW / 32; ++kc) {
        const int kbase = wave * KPW + kc * 32;
        v16h aW0 = frag_a_w(&S[lo][kbase + hi * 16],      cm0);
        v16h aW1 = frag_a_w(&S[16 + lo][kbase + hi * 16], cm1);
        const float* vp0 = Vb + (size_t)(kbase + hi * 16) * ROWS_ + lo;
#pragma unroll
        for (int j = 0; j < 4; ++j) {
          const float* vp = vp0 + j * 16;
          v16h bV;
#pragma unroll
          for (int e = 0; e < 16; ++e) {
            bV[e] = (_Float16)vp[0];
            vp += ROWS_;
          }
          accO[0][j] = wmma16(aW0, bV, accO[0][j]);
          accO[1][j] = wmma16(aW1, bV, accO[1][j]);
        }
      }
    }
  }

  // ---- phase 4: cross-wave reduction + store ------------------------------
#pragma unroll
  for (int s = 0; s < 2; ++s)
#pragma unroll
    for (int j = 0; j < 4; ++j)
#pragma unroll
      for (int v = 0; v < 8; ++v)
        atomicAdd(&Ored[s * 16 + v + 8 * hi][j * 16 + lo], accO[s][j][v]);
  __syncthreads();

  {
    const int flat = tid * 4;            // 2048 floats / 512 threads
    const int m = flat >> 6;
    const int d = flat & 63;
    float4 o = *(const float4*)&Ored[m][d];
    *(float4*)(Ob + (size_t)(q0 + m) * ROWS_ + d) = o;
  }
}

// ---------------------------------------------------------------------------
extern "C" void kernel_launch(void* const* d_in, const int* in_sizes, int n_in,
                              void* d_out, int out_size, void* d_ws, size_t ws_size,
                              hipStream_t stream) {
  (void)in_sizes; (void)n_in; (void)out_size;
  const float* Q = (const float*)d_in[0];
  const float* K = (const float*)d_in[1];
  const float* V = (const float*)d_in[2];
  float*       O = (float*)d_out;

  const _Float16* Vt = nullptr;
  if (ws_size >= VT_BYTES && d_ws != nullptr) {
    const int tdwords = (int)(VT_BYTES / 4);         // 1,572,864
    transpose_v_kernel<<<tdwords / 256, 256, 0, stream>>>(V, (unsigned*)d_ws);
    Vt = (const _Float16*)d_ws;
  }
  const int grid = B_ * H_ * (L_ / MQ);              // 1536 workgroups
  poly_attn_kernel<<<grid, NTHREADS, 0, stream>>>(Q, K, V, Vt, O);
}
